// Text2Motion_Transformer_15083925144393
// MI455X (gfx1250) — matrix operand order repaired and checked by reading, not compiled, not run on hardware
//
#include <hip/hip_runtime.h>

typedef __attribute__((ext_vector_type(16))) __bf16 v16bf;
typedef __attribute__((ext_vector_type(8)))  float  v8f;

union Frag { v16bf v; uint4 u[2]; };

#define BB     256
#define TQn    50
#define TT     51
#define CC     512
#define C4n    2048
#define MROWS  (BB*TT)      /* 13056 */
#define NHEAD  1025
#define NHEADP 1056         /* 33 x 32 */

static __device__ __forceinline__ float sigmoidf_(float x) {
  return 1.0f / (1.0f + __expf(-x));
}

// ---------------------------------------------------------------------------
// WMMA core: one wave computes a 64(M) x 32(N) f32 tile of  A[M,K] @ W[N,K]^T
// A, W are bf16 row-major with leading dimension K (multiple of 32).
// 8 WMMAs per K-step, ~121 VGPRs -> high occupancy, clean s_clause pipelining
// (the 64x64 variant hit 224 VGPRs and paid WMMA->VALU hazard nops).
// ---------------------------------------------------------------------------
static __device__ __forceinline__ void wmma_core(
    const __bf16* __restrict__ A, const __bf16* __restrict__ W,
    int K, int mBase, int nBase, v8f acc[4][2]) {
  const int lane = threadIdx.x & 31;
  const int r16  = lane & 15;
  const int aSel = (lane >> 4) << 3;   // lanes 16-31 hold K+8..15 / K+24..31
  const int bSel = (lane >> 4) << 3;   // lanes 16-31 hold K+8..15 / K+24..31
  for (int k0 = 0; k0 < K; k0 += 32) {
    Frag a[4], b[2];
#pragma unroll
    for (int mi = 0; mi < 4; ++mi) {
      const __bf16* p = A + (size_t)(mBase + mi * 16 + r16) * K + (k0 + aSel);
      a[mi].u[0] = *(const uint4*)(p);        // K = sel+0..7
      a[mi].u[1] = *(const uint4*)(p + 16);   // K = sel+16..23
    }
#pragma unroll
    for (int ni = 0; ni < 2; ++ni) {
      const __bf16* p = W + (size_t)(nBase + ni * 16 + r16) * K + (k0 + bSel);
      b[ni].u[0] = *(const uint4*)(p);        // K = sel+0..7
      b[ni].u[1] = *(const uint4*)(p + 16);   // K = sel+16..23
    }
#pragma unroll
    for (int mi = 0; mi < 4; ++mi)
#pragma unroll
      for (int ni = 0; ni < 2; ++ni)
        acc[mi][ni] = __builtin_amdgcn_wmma_f32_16x16x32_bf16(
            false, a[mi].v, false, b[ni].v, (short)0, acc[mi][ni], false, false);
  }
}

static __device__ __forceinline__ void acc_zero(v8f acc[4][2]) {
  const v8f z = {0.f, 0.f, 0.f, 0.f, 0.f, 0.f, 0.f, 0.f};
#pragma unroll
  for (int mi = 0; mi < 4; ++mi)
#pragma unroll
    for (int ni = 0; ni < 2; ++ni) acc[mi][ni] = z;
}

// C/D fragment layout: vgpr j -> row j (lanes 0-15) or row j+8 (lanes 16-31),
// column = lane & 15.
#define EPILOGUE_LOOP(BODY)                                                   \
  {                                                                           \
    const int lane = threadIdx.x & 31;                                        \
    const int rOfs = (lane >> 4) << 3;                                        \
    _Pragma("unroll") for (int mi = 0; mi < 4; ++mi)                          \
    _Pragma("unroll") for (int ni = 0; ni < 2; ++ni) {                        \
      v8f a = acc[mi][ni];                                                    \
      const int col = nBase + ni * 16 + (lane & 15);                          \
      _Pragma("unroll") for (int j = 0; j < 8; ++j) {                         \
        const int row = mBase + mi * 16 + j + rOfs;                           \
        BODY                                                                  \
      }                                                                       \
    }                                                                         \
  }

// ---- GEMM variants ---------------------------------------------------------
__global__ __launch_bounds__(32) void gemm_store_k(
    const __bf16* __restrict__ A, const __bf16* __restrict__ W,
    float* __restrict__ out, int K, int N) {
  const int mBase = blockIdx.y * 64, nBase = blockIdx.x * 32;
  v8f acc[4][2]; acc_zero(acc);
  wmma_core(A, W, K, mBase, nBase, acc);
  EPILOGUE_LOOP( out[(size_t)row * N + col] = a[j]; )
}

__global__ __launch_bounds__(32) void gemm_add_k(          // x += acc
    const __bf16* __restrict__ A, const __bf16* __restrict__ W,
    float* __restrict__ x, int K, int N) {
  const int mBase = blockIdx.y * 64, nBase = blockIdx.x * 32;
  v8f acc[4][2]; acc_zero(acc);
  wmma_core(A, W, K, mBase, nBase, acc);
  EPILOGUE_LOOP( x[(size_t)row * N + col] += a[j]; )
}

__global__ __launch_bounds__(32) void gemm_relusq_k(       // bf16(relu(acc)^2)
    const __bf16* __restrict__ A, const __bf16* __restrict__ W,
    __bf16* __restrict__ out, int K, int N) {
  const int mBase = blockIdx.y * 64, nBase = blockIdx.x * 32;
  v8f acc[4][2]; acc_zero(acc);
  wmma_core(A, W, K, mBase, nBase, acc);
  EPILOGUE_LOOP( { float r = fmaxf(a[j], 0.f);
                   out[(size_t)row * N + col] = (__bf16)(r * r); } )
}

__global__ __launch_bounds__(32) void gemm_gate_add_k(     // x += sigmoid(r)*acc
    const __bf16* __restrict__ A, const __bf16* __restrict__ W,
    const float* __restrict__ rgate, float* __restrict__ x, int K, int N) {
  const int mBase = blockIdx.y * 64, nBase = blockIdx.x * 32;
  v8f acc[4][2]; acc_zero(acc);
  wmma_core(A, W, K, mBase, nBase, acc);
  EPILOGUE_LOOP( { const size_t o = (size_t)row * N + col;
                   x[o] += sigmoidf_(rgate[o]) * a[j]; } )
}

__global__ __launch_bounds__(32) void gemm_head_k(         // col-guarded store
    const __bf16* __restrict__ A, const __bf16* __restrict__ W,
    float* __restrict__ out, int K, int Nout) {
  const int mBase = blockIdx.y * 64, nBase = blockIdx.x * 32;
  v8f acc[4][2]; acc_zero(acc);
  wmma_core(A, W, K, mBase, nBase, acc);
  EPILOGUE_LOOP( if (col < Nout) out[(size_t)row * Nout + col] = a[j]; )
}

__global__ __launch_bounds__(32) void gemm_cond_k(         // strided rows + bias
    const __bf16* __restrict__ A, const __bf16* __restrict__ W,
    const float* __restrict__ bias, float* __restrict__ x,
    int K, int N, int rowStride) {
  const int mBase = blockIdx.y * 64, nBase = blockIdx.x * 32;
  v8f acc[4][2]; acc_zero(acc);
  wmma_core(A, W, K, mBase, nBase, acc);
  EPILOGUE_LOOP( x[(size_t)row * rowStride + col] = a[j] + bias[col]; )
}

// ---------------------------------------------------------------------------
// LayerNorm helpers: one wave per row of C=512 (16 elements per lane)
// ---------------------------------------------------------------------------
static __device__ __forceinline__ void row_stats(const float v[16],
                                                 float& mu, float& rs) {
  float s = 0.f, s2 = 0.f;
#pragma unroll
  for (int i = 0; i < 16; ++i) { s += v[i]; s2 += v[i] * v[i]; }
#pragma unroll
  for (int m = 16; m >= 1; m >>= 1) {
    s  += __shfl_xor(s,  m, 32);
    s2 += __shfl_xor(s2, m, 32);
  }
  mu = s * (1.f / CC);
  float var = s2 * (1.f / CC) - mu * mu;
  rs = rsqrtf(var + 1e-5f);
}

__global__ __launch_bounds__(256) void ln_inplace_k(
    float* __restrict__ x, const float* __restrict__ g,
    const float* __restrict__ b) {
  const int wid = threadIdx.x >> 5, lane = threadIdx.x & 31;
  const int m = blockIdx.x * 8 + wid;
  float* row = x + (size_t)m * CC;
  float cur[16];
#pragma unroll
  for (int i = 0; i < 16; ++i) cur[i] = row[lane * 16 + i];
  float mu, rs; row_stats(cur, mu, rs);
#pragma unroll
  for (int i = 0; i < 16; ++i) {
    const int c = lane * 16 + i;
    row[c] = (cur[i] - mu) * rs * g[c] + b[c];
  }
}

__global__ __launch_bounds__(256) void ln_mix_att_k(
    const float* __restrict__ x, const float* __restrict__ g,
    const float* __restrict__ b, const float* __restrict__ mk,
    const float* __restrict__ mv, const float* __restrict__ mr,
    __bf16* __restrict__ xk, __bf16* __restrict__ xv, __bf16* __restrict__ xr) {
  const int wid = threadIdx.x >> 5, lane = threadIdx.x & 31;
  const int m = blockIdx.x * 8 + wid;
  const int t = m % TT;
  const float* row = x + (size_t)m * CC;
  float cur[16];
#pragma unroll
  for (int i = 0; i < 16; ++i) cur[i] = row[lane * 16 + i];
  float mu, rs; row_stats(cur, mu, rs);
  float prv[16]; float mup = 0.f, rsp = 0.f;
  if (t > 0) {
#pragma unroll
    for (int i = 0; i < 16; ++i) prv[i] = row[lane * 16 + i - CC];
    row_stats(prv, mup, rsp);
  }
#pragma unroll
  for (int i = 0; i < 16; ++i) {
    const int c = lane * 16 + i;
    const float h  = (cur[i] - mu) * rs * g[c] + b[c];
    const float hp = (t > 0) ? ((prv[i] - mup) * rsp * g[c] + b[c]) : 0.f;
    const size_t o = (size_t)m * CC + c;
    const float a_k = mk[c], a_v = mv[c], a_r = mr[c];
    xk[o] = (__bf16)(h * a_k + hp * (1.f - a_k));
    xv[o] = (__bf16)(h * a_v + hp * (1.f - a_v));
    xr[o] = (__bf16)(h * a_r + hp * (1.f - a_r));
  }
}

__global__ __launch_bounds__(256) void ln_mix_ffn_k(
    const float* __restrict__ x, const float* __restrict__ g,
    const float* __restrict__ b, const float* __restrict__ mk,
    const float* __restrict__ mr,
    __bf16* __restrict__ xk, __bf16* __restrict__ xr) {
  const int wid = threadIdx.x >> 5, lane = threadIdx.x & 31;
  const int m = blockIdx.x * 8 + wid;
  const int t = m % TT;
  const float* row = x + (size_t)m * CC;
  float cur[16];
#pragma unroll
  for (int i = 0; i < 16; ++i) cur[i] = row[lane * 16 + i];
  float mu, rs; row_stats(cur, mu, rs);
  float prv[16]; float mup = 0.f, rsp = 0.f;
  if (t > 0) {
#pragma unroll
    for (int i = 0; i < 16; ++i) prv[i] = row[lane * 16 + i - CC];
    row_stats(prv, mup, rsp);
  }
#pragma unroll
  for (int i = 0; i < 16; ++i) {
    const int c = lane * 16 + i;
    const float h  = (cur[i] - mu) * rs * g[c] + b[c];
    const float hp = (t > 0) ? ((prv[i] - mup) * rsp * g[c] + b[c]) : 0.f;
    const size_t o = (size_t)m * CC + c;
    const float a_k = mk[c], a_r = mr[c];
    xk[o] = (__bf16)(h * a_k + hp * (1.f - a_k));
    xr[o] = (__bf16)(h * a_r + hp * (1.f - a_r));
  }
}

__global__ __launch_bounds__(256) void lnf_bf16_k(
    const float* __restrict__ x, const float* __restrict__ g,
    const float* __restrict__ b, __bf16* __restrict__ out) {
  const int wid = threadIdx.x >> 5, lane = threadIdx.x & 31;
  const int m = blockIdx.x * 8 + wid;
  const float* row = x + (size_t)m * CC;
  float cur[16];
#pragma unroll
  for (int i = 0; i < 16; ++i) cur[i] = row[lane * 16 + i];
  float mu, rs; row_stats(cur, mu, rs);
#pragma unroll
  for (int i = 0; i < 16; ++i) {
    const int c = lane * 16 + i;
    out[(size_t)m * CC + c] = (__bf16)((cur[i] - mu) * rs * g[c] + b[c]);
  }
}

// ---------------------------------------------------------------------------
// WKV scan (numerically-stable RWKV v4), fused with sigmoid(r) gating.
// One thread per (b, c); sequential over t.
// ---------------------------------------------------------------------------
__global__ __launch_bounds__(256) void wkv_k(
    const float* __restrict__ kbuf, const float* __restrict__ vbuf,
    const float* __restrict__ rbuf, const float* __restrict__ tdecay,
    const float* __restrict__ tfirst, __bf16* __restrict__ out) {
  const int tid = blockIdx.x * 256 + threadIdx.x;   // b*CC + c
  const int c = tid & (CC - 1);
  const int bidx = tid >> 9;
  const float w = -__expf(tdecay[c]);
  const float u = tfirst[c];
  float aa = 0.f, bb = 0.f, pp = -1e38f;
  for (int t = 0; t < TT; ++t) {
    const size_t o = ((size_t)bidx * TT + t) * CC + c;
    const float kt = kbuf[o], vt = vbuf[o];
    const float ww = u + kt;
    const float p  = fmaxf(pp, ww);
    const float e1 = __expf(pp - p), e2 = __expf(ww - p);
    const float y  = (e1 * aa + e2 * vt) / (e1 * bb + e2);
    out[o] = (__bf16)(y * sigmoidf_(rbuf[o]));
    const float ww2 = pp + w;
    const float p2  = fmaxf(ww2, kt);
    const float e1n = __expf(ww2 - p2), e2n = __expf(kt - p2);
    aa = e1n * aa + e2n * vt;
    bb = e1n * bb + e2n;
    pp = p2;
  }
}

// ---------------------------------------------------------------------------
// Embedding gather + conversions
// ---------------------------------------------------------------------------
__global__ __launch_bounds__(256) void embed_k(
    const int* __restrict__ idxs, const float* __restrict__ emb,
    float* __restrict__ x) {
  const size_t tid = (size_t)blockIdx.x * 256 + threadIdx.x; // B*TQ*C threads
  const int c = (int)(tid & (CC - 1));
  const size_t bt = tid >> 9;
  const int t = (int)(bt % TQn);
  const int bidx = (int)(bt / TQn);
  const int tok = idxs[bidx * TQn + t];
  x[((size_t)bidx * TT + t + 1) * CC + c] = emb[(size_t)tok * CC + c];
}

__global__ __launch_bounds__(256) void convert_k(
    const float* __restrict__ in, __bf16* __restrict__ out, int n) {
  const int i = blockIdx.x * 256 + threadIdx.x;
  if (i < n) out[i] = (__bf16)in[i];
}

__global__ __launch_bounds__(256) void convert_padrows_k(
    const float* __restrict__ in, __bf16* __restrict__ out,
    int srcRows, int dstRows, int cols) {
  const int i = blockIdx.x * 256 + threadIdx.x;
  if (i >= dstRows * cols) return;
  const int r = i / cols;
  out[i] = (r < srcRows) ? (__bf16)in[i] : (__bf16)0.f;
}

// ---------------------------------------------------------------------------
// Host launcher
// ---------------------------------------------------------------------------
extern "C" void kernel_launch(void* const* d_in, const int* in_sizes, int n_in,
                              void* d_out, int out_size, void* d_ws, size_t ws_size,
                              hipStream_t stream) {
  (void)in_sizes; (void)n_in; (void)out_size; (void)ws_size;
  // ---- input pointers (idxs, clip, then params flattened jax-style) -------
  const int*   idxs = (const int*)d_in[0];
  const float* clip = (const float*)d_in[1];
  // blocks[0] @ 2 (20 leaves incl ln0), blocks[1] @ 22 (18 leaves)
  const float *att_k[2], *att_mix_k[2], *att_mix_r[2], *att_mix_v[2];
  const float *att_o[2], *att_r[2], *att_v[2];
  const float *ffn_k[2], *ffn_mix_k[2], *ffn_mix_r[2], *ffn_r[2], *ffn_v[2];
  const float *ln1_b[2], *ln1_g[2], *ln2_b[2], *ln2_g[2], *tdec[2], *tfst[2];
  {
    int base = 2;
    att_k[0]=(const float*)d_in[base+0]; att_mix_k[0]=(const float*)d_in[base+1];
    att_mix_r[0]=(const float*)d_in[base+2]; att_mix_v[0]=(const float*)d_in[base+3];
    att_o[0]=(const float*)d_in[base+4]; att_r[0]=(const float*)d_in[base+5];
    att_v[0]=(const float*)d_in[base+6]; ffn_k[0]=(const float*)d_in[base+7];
    ffn_mix_k[0]=(const float*)d_in[base+8]; ffn_mix_r[0]=(const float*)d_in[base+9];
    ffn_r[0]=(const float*)d_in[base+10]; ffn_v[0]=(const float*)d_in[base+11];
    // base+12 = ln0_b, base+13 = ln0_g
    ln1_b[0]=(const float*)d_in[base+14]; ln1_g[0]=(const float*)d_in[base+15];
    ln2_b[0]=(const float*)d_in[base+16]; ln2_g[0]=(const float*)d_in[base+17];
    tdec[0]=(const float*)d_in[base+18]; tfst[0]=(const float*)d_in[base+19];
  }
  const float* ln0_b = (const float*)d_in[14];
  const float* ln0_g = (const float*)d_in[15];
  {
    int base = 22;
    att_k[1]=(const float*)d_in[base+0]; att_mix_k[1]=(const float*)d_in[base+1];
    att_mix_r[1]=(const float*)d_in[base+2]; att_mix_v[1]=(const float*)d_in[base+3];
    att_o[1]=(const float*)d_in[base+4]; att_r[1]=(const float*)d_in[base+5];
    att_v[1]=(const float*)d_in[base+6]; ffn_k[1]=(const float*)d_in[base+7];
    ffn_mix_k[1]=(const float*)d_in[base+8]; ffn_mix_r[1]=(const float*)d_in[base+9];
    ffn_r[1]=(const float*)d_in[base+10]; ffn_v[1]=(const float*)d_in[base+11];
    ln1_b[1]=(const float*)d_in[base+12]; ln1_g[1]=(const float*)d_in[base+13];
    ln2_b[1]=(const float*)d_in[base+14]; ln2_g[1]=(const float*)d_in[base+15];
    tdec[1]=(const float*)d_in[base+16]; tfst[1]=(const float*)d_in[base+17];
  }
  const float* cond_b  = (const float*)d_in[40];
  const float* cond_w  = (const float*)d_in[41];
  const float* head_w  = (const float*)d_in[42];
  const float* lnf_b   = (const float*)d_in[43];
  const float* lnf_g   = (const float*)d_in[44];
  const float* tok_emb = (const float*)d_in[45];
  float* out = (float*)d_out;

  // ---- workspace layout ---------------------------------------------------
  size_t off = 0;
  auto alloc = [&](size_t bytes) -> void* {
    void* p = (char*)d_ws + off;
    off += (bytes + 255) & ~(size_t)255;
    return p;
  };
  const size_t MC = (size_t)MROWS * CC;
  float*  xf32  = (float*)alloc(MC * 4);
  float*  kf32  = (float*)alloc(MC * 4);
  float*  vf32  = (float*)alloc(MC * 4);
  float*  rf32  = (float*)alloc(MC * 4);
  __bf16* bufA  = (__bf16*)alloc(MC * 2);
  __bf16* bufB  = (__bf16*)alloc(MC * 2);
  __bf16* bufC  = (__bf16*)alloc(MC * 2);
  __bf16* kkbf  = (__bf16*)alloc((size_t)MROWS * C4n * 2);
  __bf16 *wb_attk[2], *wb_attv[2], *wb_attr[2], *wb_atto[2];
  __bf16 *wb_ffnk[2], *wb_ffnv[2], *wb_ffnr[2];
  for (int l = 0; l < 2; ++l) {
    wb_attk[l] = (__bf16*)alloc((size_t)CC * CC * 2);
    wb_attv[l] = (__bf16*)alloc((size_t)CC * CC * 2);
    wb_attr[l] = (__bf16*)alloc((size_t)CC * CC * 2);
    wb_atto[l] = (__bf16*)alloc((size_t)CC * CC * 2);
    wb_ffnk[l] = (__bf16*)alloc((size_t)C4n * CC * 2);
    wb_ffnv[l] = (__bf16*)alloc((size_t)CC * C4n * 2);
    wb_ffnr[l] = (__bf16*)alloc((size_t)CC * CC * 2);
  }
  __bf16* wb_cond = (__bf16*)alloc((size_t)CC * CC * 2);
  __bf16* wb_head = (__bf16*)alloc((size_t)NHEADP * CC * 2);
  __bf16* clip_bf = (__bf16*)alloc((size_t)BB * CC * 2);

  // ---- weight / input conversions ----------------------------------------
  auto conv = [&](const float* src, __bf16* dst, int n) {
    convert_k<<<(n + 255) / 256, 256, 0, stream>>>(src, dst, n);
  };
  for (int l = 0; l < 2; ++l) {
    conv(att_k[l], wb_attk[l], CC * CC);
    conv(att_v[l], wb_attv[l], CC * CC);
    conv(att_r[l], wb_attr[l], CC * CC);
    conv(att_o[l], wb_atto[l], CC * CC);
    conv(ffn_k[l], wb_ffnk[l], C4n * CC);
    conv(ffn_v[l], wb_ffnv[l], CC * C4n);
    conv(ffn_r[l], wb_ffnr[l], CC * CC);
  }
  conv(cond_w, wb_cond, CC * CC);
  conv(clip, clip_bf, BB * CC);
  convert_padrows_k<<<(NHEADP * CC + 255) / 256, 256, 0, stream>>>(
      head_w, wb_head, NHEAD, NHEADP, CC);

  // ---- assemble x = [cond ; tok_emb[idxs]] --------------------------------
  embed_k<<<(BB * TQn * CC) / 256, 256, 0, stream>>>(idxs, tok_emb, xf32);
  gemm_cond_k<<<dim3(CC / 32, BB / 64), 32, 0, stream>>>(
      clip_bf, wb_cond, cond_b, xf32, CC, CC, TT * CC);

  const dim3 gN512(CC / 32, MROWS / 64);       // 16 x 204
  const dim3 gN2048(C4n / 32, MROWS / 64);     // 64 x 204
  const dim3 gHead(NHEADP / 32, MROWS / 64);   // 33 x 204
  const int lnBlocks = MROWS / 8;              // 1632

  // ln0 (layer 0 only)
  ln_inplace_k<<<lnBlocks, 256, 0, stream>>>(xf32, ln0_g, ln0_b);

  for (int l = 0; l < 2; ++l) {
    // ---- time-mix ----
    ln_mix_att_k<<<lnBlocks, 256, 0, stream>>>(
        xf32, ln1_g[l], ln1_b[l], att_mix_k[l], att_mix_v[l], att_mix_r[l],
        bufA, bufB, bufC);
    gemm_store_k<<<gN512, 32, 0, stream>>>(bufA, wb_attk[l], kf32, CC, CC);
    gemm_store_k<<<gN512, 32, 0, stream>>>(bufB, wb_attv[l], vf32, CC, CC);
    gemm_store_k<<<gN512, 32, 0, stream>>>(bufC, wb_attr[l], rf32, CC, CC);
    wkv_k<<<(BB * CC) / 256, 256, 0, stream>>>(
        kf32, vf32, rf32, tdec[l], tfst[l], bufA);
    gemm_add_k<<<gN512, 32, 0, stream>>>(bufA, wb_atto[l], xf32, CC, CC);
    // ---- channel-mix ----
    ln_mix_ffn_k<<<lnBlocks, 256, 0, stream>>>(
        xf32, ln2_g[l], ln2_b[l], ffn_mix_k[l], ffn_mix_r[l], bufA, bufC);
    gemm_relusq_k<<<gN2048, 32, 0, stream>>>(bufA, wb_ffnk[l], kkbf, CC, C4n);
    gemm_store_k<<<gN512, 32, 0, stream>>>(bufC, wb_ffnr[l], rf32, CC, CC);
    gemm_gate_add_k<<<gN512, 32, 0, stream>>>(kkbf, wb_ffnv[l], rf32, xf32,
                                              C4n, CC);
  }

  // ---- final LN + head ----------------------------------------------------
  lnf_bf16_k<<<lnBlocks, 256, 0, stream>>>(xf32, lnf_g, lnf_b, bufA);
  gemm_head_k<<<gHead, 32, 0, stream>>>(bufA, wb_head, out, CC, NHEAD);
}
